// Decoder_25434796327250
// MI455X (gfx1250) — matrix operand order repaired and checked by reading, not compiled, not run on hardware
//
#include <hip/hip_runtime.h>
#include <hip/hip_bf16.h>

// ---------------------------------------------------------------------------
// CDNA5 (gfx1250) decoder: bf16 WMMA GEMMs with TDM (tensor_load_to_lds)
// double-buffered LDS staging of the activation strip, fused attention/GRU,
// in-place log-softmax. Weights converted fp32->bf16 once per call (74 MB,
// resident in the 192 MB L2 across the 63 recurrence steps).
// ---------------------------------------------------------------------------

typedef __attribute__((ext_vector_type(16))) __bf16 v16bf;
typedef __attribute__((ext_vector_type(8)))  __bf16 v8bf;
typedef __attribute__((ext_vector_type(8)))  float  v8f;
typedef __attribute__((ext_vector_type(4)))  unsigned int u32x4;
typedef __attribute__((ext_vector_type(8)))  int i32x8;
typedef __attribute__((ext_vector_type(4)))  int i32x4;

static constexpr int Bc = 32;     // batch
static constexpr int Sc = 48;     // ctx seq
static constexpr int Tc = 64;     // decode steps
static constexpr int Hc = 1024;   // hidden
static constexpr int Cc = 2048;   // ctx dim
static constexpr int Ec = 512;    // embed dim
static constexpr int Vc = 32000;  // vocab

// ---------------------------------------------------------------- converts --
__global__ void k_f32_to_bf16(const float* __restrict__ src,
                              __bf16* __restrict__ dst, int n) {
  int i = blockIdx.x * 256 + threadIdx.x;
  if (i < n) dst[i] = (__bf16)src[i];
}

// embeddings[:, :T-1, :]  ->  contiguous bf16 (B*(T-1)) x E
__global__ void k_gather_emb(const float* __restrict__ emb,
                             __bf16* __restrict__ dst) {
  int i = blockIdx.x * 256 + threadIdx.x;
  const int n = Bc * (Tc - 1) * Ec;
  if (i >= n) return;
  int f   = i % Ec;
  int row = i / Ec;
  int b   = row / (Tc - 1);
  int t   = row % (Tc - 1);
  dst[i] = (__bf16)emb[((size_t)b * Tc + t) * Ec + f];
}

__global__ void k_init_h(const float* __restrict__ h0,
                         float* __restrict__ hr, float* __restrict__ hl,
                         __bf16* __restrict__ hrb, __bf16* __restrict__ hlb) {
  int i = blockIdx.x * 256 + threadIdx.x;
  if (i >= Bc * Hc) return;
  float v = h0[i];
  hr[i] = v; hl[i] = v;
  hrb[i] = (__bf16)v; hlb[i] = (__bf16)v;
}

__global__ void k_init_full0(const float* __restrict__ out0,
                             float* __restrict__ full) {
  int i = blockIdx.x * 256 + threadIdx.x;
  if (i >= Bc * Ec) return;
  int b = i / Ec, f = i % Ec;
  full[(size_t)b * Tc * Ec + f] = out0[i];
}

// full[:, 1:, :] += emb_proj  (emb_proj rows are (b, t), t in [0, T-2])
__global__ void k_add_emb(const float* __restrict__ ep,
                          float* __restrict__ full) {
  int i = blockIdx.x * 256 + threadIdx.x;
  const int n = Bc * (Tc - 1) * Ec;
  if (i >= n) return;
  int f   = i % Ec;
  int row = i / Ec;
  int b   = row / (Tc - 1);
  int t   = row % (Tc - 1);
  full[((size_t)b * Tc + (t + 1)) * Ec + f] += ep[i];
}

// ------------------------------------------------- TDM descriptor + issue ---
// Issue a TENSOR_LOAD_TO_LDS of a 2D tile: R rows x KC bf16 cols from X
// starting at element column col0, row stride K; destination LDS offset.
// D# layout per CDNA5 ISA ch.8 (group0: count/lds/global/type, group1: dims).
__device__ __forceinline__ void tdm_load_strip(const __bf16* gsrc,
                                               unsigned int lds_off,
                                               int K, int R, int KC) {
  unsigned long long ga = (unsigned long long)(uintptr_t)gsrc;
  u32x4 g0;
  g0[0] = 1u;                                   // count=1, user descriptor
  g0[1] = lds_off;                              // LDS destination address
  g0[2] = (unsigned int)ga;                     // global_addr[31:0]
  g0[3] = (unsigned int)(ga >> 32) | (2u << 30);// global_addr[56:32], type=2
  i32x8 g1;
  g1[0] = 1 << 16;                              // data_size=1 (2 bytes)
  g1[1] = (K & 0xffff) << 16;                   // tensor_dim0[15:0]
  g1[2] = ((K >> 16) & 0xffff) | ((R & 0xffff) << 16);  // dim0 hi | dim1 lo
  g1[3] = (KC & 0xffff) << 16;                  // dim1 hi=0 | tile_dim0=KC
  g1[4] = R & 0xffff;                           // tile_dim1=R, tile_dim2=0
  g1[5] = K;                                    // tensor_dim0_stride[31:0]
  g1[6] = 0;                                    // stride hi / dim1_stride lo
  g1[7] = 0;
  i32x4 gz = (i32x4)0;                          // groups 2/3 unused (2D)
#if defined(__clang_major__) && __clang_major__ >= 23
  __builtin_amdgcn_tensor_load_to_lds(g0, g1, gz, gz, (i32x8)0, 0);
#else
  __builtin_amdgcn_tensor_load_to_lds(g0, g1, gz, gz, 0);
#endif
}

// ------------------------------------------------------------ WMMA GEMM ----
// out[n, f] = sum_k X[n, k] * W[f, k]  (+ bias[f])
// W: M x K bf16 row-major (streamed from global/L2), X: N x K bf16 row-major
// (strip of R = NT*16 rows staged into LDS by TDM, double-buffered K-chunks).
// Requirements: M/16 % 8 == 0, (N/16) % NT == 0, K % KC == 0.
template <int NT>
__global__ void k_gemm_bf16(const __bf16* __restrict__ W,
                            const __bf16* __restrict__ X,
                            const float* __restrict__ bias,
                            float* __restrict__ out,
                            int M, int N, int K, long ostride) {
  constexpr int R  = NT * 16;       // staged X rows per block
  constexpr int KC = 16384 / R;     // chunk cols: R*KC*2B = 32 KB per buffer
  extern __shared__ char smem[];    // 2 * 32 KB double buffer
  __bf16* sX = (__bf16*)smem;

  const int lane = threadIdx.x & 31;
  const int wave = threadIdx.x >> 5;
  const int lo   = lane & 15;       // C/D column (N); A row (M) within tile
  const int hi   = lane >> 4;       // half-wave selector for K chunks / M rows

  const int mblocks = (M >> 4) >> 3;              // m-tiles / 8 waves
  const int ng = blockIdx.x / mblocks;            // shared by all 8 waves
  const int mb = blockIdx.x % mblocks;
  const int m0 = (mb * 8 + wave) * 16;
  const int n0 = ng * R;

  const int nchunks = K / KC;
  const unsigned int lds_base =
      (unsigned int)(uintptr_t)smem;              // low 32 bits = LDS offset

  // Kick off chunk 0 DMA (one wave per block issues; EXEC ignored by TDM).
  if (wave == 0)
    tdm_load_strip(X + (size_t)n0 * K, lds_base, K, R, KC);

  v8f acc[NT];
#pragma unroll
  for (int t = 0; t < NT; ++t)
#pragma unroll
    for (int r = 0; r < 8; ++r) acc[t][r] = 0.0f;

  // A tile rows: m = m0 + lo; lanes 0-15 take K = [kb+0..7, kb+16..23],
  // lanes 16-31 take K = [kb+8..15, kb+24..31]  (ISA 7.12.2, 16-bit A 16x32).
  const __bf16* wrow = W + (size_t)(m0 + lo) * K;

  for (int c = 0; c < nchunks; ++c) {
    if (wave == 0) {
      if (c + 1 < nchunks) {  // overlap DMA of next chunk with compute
        tdm_load_strip(X + (size_t)n0 * K + (c + 1) * KC,
                       lds_base + (unsigned int)(((c + 1) & 1) * R * KC * 2),
                       K, R, KC);
        __builtin_amdgcn_s_wait_tensorcnt(1);  // chunk c finished
      } else {
        __builtin_amdgcn_s_wait_tensorcnt(0);
      }
    }
    __syncthreads();  // staged chunk c visible to all waves

    const __bf16* sb = sX + (size_t)(c & 1) * R * KC;
    const int kbase = c * KC;
#pragma unroll 4
    for (int kb = 0; kb < KC; kb += 32) {
      if (kbase + kb + 256 < K)
        __builtin_prefetch(wrow + kbase + kb + 256, 0, 1);  // stream weights

      union { v16bf v; v8bf h[2]; } A;
      A.h[0] = *reinterpret_cast<const v8bf*>(wrow + kbase + kb + 8 * hi);
      A.h[1] = *reinterpret_cast<const v8bf*>(wrow + kbase + kb + 16 + 8 * hi);

#pragma unroll
      for (int t = 0; t < NT; ++t) {
        // B tile: lane holds K = [kb+8*hi..+7, kb+16+8*hi..+7] of LDS row n.
        const __bf16* bp = sb + (size_t)(t * 16 + lo) * KC + kb + 8 * hi;
        union { v16bf v; v8bf h[2]; } Bv;
        Bv.h[0] = *reinterpret_cast<const v8bf*>(bp);
        Bv.h[1] = *reinterpret_cast<const v8bf*>(bp + 16);
        acc[t] = __builtin_amdgcn_wmma_f32_16x16x32_bf16(
            false, A.v, false, Bv.v, (short)0, acc[t], false, false);
      }
    }
    __syncthreads();  // everyone done with buffer before TDM overwrites it
  }

  // C/D layout: lane -> N = lo, VGPR r -> M = m0 + 8*hi + r
  const int fbase = m0 + 8 * hi;
#pragma unroll
  for (int t = 0; t < NT; ++t) {
    const int n = n0 + t * 16 + lo;
    float* o = out + (size_t)n * ostride + fbase;
#pragma unroll
    for (int r = 0; r < 8; ++r) {
      float v = acc[t][r];
      if (bias) v += bias[fbase + r];
      o[r] = v;
    }
  }
}

// ------------------------------------------------------------- attention ---
// One workgroup (256 thr = 8 waves) per batch row.
// e_s = v . tanh(enc[b,s,:] + q[b,:]); softmax over S; x = sum_s w_s ctx[b,s,:]
__global__ void k_attn(const float* __restrict__ q,
                       const float* __restrict__ enc,
                       const float* __restrict__ v_att,
                       const float* __restrict__ mask,
                       const float* __restrict__ ctx,
                       __bf16* __restrict__ x_b) {
  const int b    = blockIdx.x;
  const int tid  = threadIdx.x;
  const int lane = tid & 31;
  const int wave = tid >> 5;
  __shared__ float sE[Sc];
  __shared__ float sW[Sc];

  const float* qb = q + (size_t)b * Hc;
  for (int s = wave; s < Sc; s += 8) {
    const float* er = enc + ((size_t)b * Sc + s) * Hc;
    float part = 0.0f;
    for (int j = lane; j < Hc; j += 32)
      part += v_att[j] * tanhf(er[j] + qb[j]);
#pragma unroll
    for (int off = 16; off >= 1; off >>= 1)
      part += __shfl_xor(part, off, 32);
    if (lane == 0)
      sE[s] = (mask[b * Sc + s] > 0.5f) ? -1.0e9f : part;
  }
  __syncthreads();

  float mx = -3.4e38f;
  for (int s = 0; s < Sc; ++s) mx = fmaxf(mx, sE[s]);
  float sum = 0.0f;
  for (int s = 0; s < Sc; ++s) sum += expf(sE[s] - mx);
  if (tid < Sc) sW[tid] = expf(sE[tid] - mx) / sum;
  __syncthreads();

  for (int c = tid; c < Cc; c += 256) {
    float acc = 0.0f;
    const float* cb = ctx + (size_t)b * Sc * Cc + c;
    for (int s = 0; s < Sc; ++s) acc += sW[s] * cb[(size_t)s * Cc];
    x_b[(size_t)b * Cc + c] = (__bf16)acc;
  }
}

// ----------------------------------------------------------------- gates ---
// h' = (1-z)*n + z*h with r = sig(ir+hr), z = sig(iz+hz), n = tanh(in + r*hn)
__global__ void k_gates(const float* __restrict__ gi,
                        const float* __restrict__ gh,
                        float* __restrict__ h, __bf16* __restrict__ h_b) {
  int i = blockIdx.x * 256 + threadIdx.x;
  if (i >= Bc * Hc) return;
  int b = i >> 10, j = i & (Hc - 1);
  const float* gib = gi + (size_t)b * 3 * Hc;
  const float* ghb = gh + (size_t)b * 3 * Hc;
  float r = 1.0f / (1.0f + expf(-(gib[j] + ghb[j])));
  float z = 1.0f / (1.0f + expf(-(gib[Hc + j] + ghb[Hc + j])));
  float n = tanhf(gib[2 * Hc + j] + r * ghb[2 * Hc + j]);
  float hv = (1.0f - z) * n + z * h[i];
  h[i] = hv;
  h_b[i] = (__bf16)hv;
}

// ----------------------------------------------------------- log-softmax ---
// In-place per-row log-softmax on (B*T) x V fp32.
__global__ void k_logsoftmax(float* __restrict__ x) {
  __shared__ float red[256];
  float* row = x + (size_t)blockIdx.x * Vc;
  const int tid = threadIdx.x;

  float mx = -3.4e38f;
  for (int i = tid; i < Vc; i += 256) mx = fmaxf(mx, row[i]);
  red[tid] = mx;
  __syncthreads();
  for (int s = 128; s > 0; s >>= 1) {
    if (tid < s) red[tid] = fmaxf(red[tid], red[tid + s]);
    __syncthreads();
  }
  mx = red[0];
  __syncthreads();

  float sum = 0.0f;
  for (int i = tid; i < Vc; i += 256) sum += expf(row[i] - mx);
  red[tid] = sum;
  __syncthreads();
  for (int s = 128; s > 0; s >>= 1) {
    if (tid < s) red[tid] += red[tid + s];
    __syncthreads();
  }
  const float lse = mx + logf(red[0]);
  for (int i = tid; i < Vc; i += 256) row[i] = row[i] - lse;
}

// ---------------------------------------------------------------- driver ---
extern "C" void kernel_launch(void* const* d_in, const int* in_sizes, int n_in,
                              void* d_out, int out_size, void* d_ws,
                              size_t ws_size, hipStream_t stream) {
  const float* ctx      = (const float*)d_in[0];
  const float* ctx_mask = (const float*)d_in[1];
  const float* embeds   = (const float*)d_in[2];
  const float* hiddens  = (const float*)d_in[3];
  const float* enc      = (const float*)d_in[4];
  const float* out0     = (const float*)d_in[5];
  const float* Wq       = (const float*)d_in[6];
  const float* bq       = (const float*)d_in[7];
  const float* v_att    = (const float*)d_in[8];
  const float* Wih_r    = (const float*)d_in[9];
  const float* Whh_r    = (const float*)d_in[10];
  const float* bih_r    = (const float*)d_in[11];
  const float* bhh_r    = (const float*)d_in[12];
  const float* Wih_l    = (const float*)d_in[13];
  const float* Whh_l    = (const float*)d_in[14];
  const float* bih_l    = (const float*)d_in[15];
  const float* bhh_l    = (const float*)d_in[16];
  const float* Wout     = (const float*)d_in[17];
  const float* bout     = (const float*)d_in[18];
  const float* Wemb     = (const float*)d_in[19];
  const float* bemb     = (const float*)d_in[20];
  const float* Wgen     = (const float*)d_in[21];

  // ---- workspace carve-up (256B aligned) ----
  uint8_t* ws = (uint8_t*)d_ws;
  size_t off = 0;
  auto alloc = [&](size_t bytes) -> void* {
    void* p = ws + off;
    off += (bytes + 255) & ~(size_t)255;
    return p;
  };
  __bf16* Wq_b    = (__bf16*)alloc((size_t)Hc * Hc * 2);
  __bf16* Whh_r_b = (__bf16*)alloc((size_t)3 * Hc * Hc * 2);
  __bf16* Whh_l_b = (__bf16*)alloc((size_t)3 * Hc * Hc * 2);
  __bf16* Wih_r_b = (__bf16*)alloc((size_t)3 * Hc * Cc * 2);
  __bf16* Wih_l_b = (__bf16*)alloc((size_t)3 * Hc * Cc * 2);
  __bf16* Wout_b  = (__bf16*)alloc((size_t)Ec * Hc * 2);
  __bf16* Wemb_b  = (__bf16*)alloc((size_t)Ec * Ec * 2);
  __bf16* Wgen_b  = (__bf16*)alloc((size_t)Vc * Ec * 2);
  float*  h_r     = (float*)alloc((size_t)Bc * Hc * 4);
  float*  h_l     = (float*)alloc((size_t)Bc * Hc * 4);
  __bf16* h_r_b   = (__bf16*)alloc((size_t)Bc * Hc * 2);
  __bf16* h_l_b   = (__bf16*)alloc((size_t)Bc * Hc * 2);
  float*  q_f     = (float*)alloc((size_t)Bc * Hc * 4);
  float*  gh_f    = (float*)alloc((size_t)Bc * 3 * Hc * 4);
  float*  gi_f    = (float*)alloc((size_t)Bc * 3 * Hc * 4);
  __bf16* x_b     = (__bf16*)alloc((size_t)Bc * Cc * 2);
  __bf16* emb_c   = (__bf16*)alloc((size_t)Bc * (Tc - 1) * Ec * 2);
  float*  emb_p   = (float*)alloc((size_t)Bc * (Tc - 1) * Ec * 4);
  float*  full    = (float*)alloc((size_t)Bc * Tc * Ec * 4);
  __bf16* full_b  = (__bf16*)alloc((size_t)Bc * Tc * Ec * 2);
  (void)ws_size; (void)n_in; (void)in_sizes; (void)out_size;

  auto cvt = [&](const float* s, __bf16* d, int n) {
    k_f32_to_bf16<<<(n + 255) / 256, 256, 0, stream>>>(s, d, n);
  };
  // Requirements: (M/16)%8==0, (N/16)%NT==0, K % (16384/(NT*16)) == 0.
  constexpr int kLdsBytes = 2 * 32768;  // double-buffered 32 KB strips
  auto gemm = [&](int NT, const __bf16* W, const __bf16* X, const float* bias,
                  float* o, int M, int N, int K, long ostride) {
    const int mblocks = (M >> 4) >> 3;
    const int blocks  = mblocks * ((N >> 4) / NT);
    if (NT == 2)
      k_gemm_bf16<2><<<blocks, 256, kLdsBytes, stream>>>(W, X, bias, o,
                                                         M, N, K, ostride);
    else
      k_gemm_bf16<4><<<blocks, 256, kLdsBytes, stream>>>(W, X, bias, o,
                                                         M, N, K, ostride);
  };

  // ---- weight conversion (bf16 copies live in L2 for the recurrence) ----
  cvt(Wq, Wq_b, Hc * Hc);
  cvt(Whh_r, Whh_r_b, 3 * Hc * Hc);
  cvt(Whh_l, Whh_l_b, 3 * Hc * Hc);
  cvt(Wih_r, Wih_r_b, 3 * Hc * Cc);
  cvt(Wih_l, Wih_l_b, 3 * Hc * Cc);
  cvt(Wout, Wout_b, Ec * Hc);
  cvt(Wemb, Wemb_b, Ec * Ec);
  cvt(Wgen, Wgen_b, Vc * Ec);

  // ---- init ----
  k_init_h<<<(Bc * Hc + 255) / 256, 256, 0, stream>>>(hiddens, h_r, h_l,
                                                      h_r_b, h_l_b);
  k_init_full0<<<(Bc * Ec + 255) / 256, 256, 0, stream>>>(out0, full);
  k_gather_emb<<<(Bc * (Tc - 1) * Ec + 255) / 256, 256, 0, stream>>>(embeds,
                                                                     emb_c);

  // ---- recurrence: one step ----
  auto step = [&](float* h, __bf16* hb, const __bf16* WihB, const __bf16* WhhB,
                  const float* bih, const float* bhh, int full_col) {
    gemm(2, Wq_b, hb, bq, q_f, Hc, Bc, Hc, Hc);                 // q = h Wq^T+bq
    gemm(2, WhhB, hb, bhh, gh_f, 3 * Hc, Bc, Hc, 3 * Hc);       // gh
    k_attn<<<Bc, 256, 0, stream>>>(q_f, enc, v_att, ctx_mask, ctx, x_b);
    gemm(2, WihB, x_b, bih, gi_f, 3 * Hc, Bc, Cc, 3 * Hc);      // gi
    k_gates<<<(Bc * Hc + 255) / 256, 256, 0, stream>>>(gi_f, gh_f, h, hb);
    gemm(2, Wout_b, hb, bout, full + (size_t)full_col * Ec,     // out_t
         Ec, Bc, Hc, (long)Tc * Ec);
  };

  const int n_right = Tc / 2;        // 32 steps -> full cols 1,3,...,63
  const int n_left  = (Tc - 1) / 2;  // 31 steps -> full cols 2,4,...,62
  for (int t = 0; t < n_right; ++t)
    step(h_r, h_r_b, Wih_r_b, Whh_r_b, bih_r, bhh_r, 1 + 2 * t);
  for (int t = 0; t < n_left; ++t)
    step(h_l, h_l_b, Wih_l_b, Whh_l_b, bih_l, bhh_l, 2 + 2 * t);

  // ---- embedding projection and merge ----
  gemm(2, Wemb_b, emb_c, bemb, emb_p, Ec, Bc * (Tc - 1), Ec, Ec);
  k_add_emb<<<(Bc * (Tc - 1) * Ec + 255) / 256, 256, 0, stream>>>(emb_p, full);
  cvt(full, full_b, Bc * Tc * Ec);

  // ---- logits (dominant GEMM: 32000 x 2048 x 512) straight into d_out ----
  gemm(4, Wgen_b, full_b, nullptr, (float*)d_out, Vc, Bc * Tc, Ec, Vc);

  // ---- in-place log-softmax over vocab ----
  k_logsoftmax<<<Bc * Tc, 256, 0, stream>>>((float*)d_out);
}
